// GroupedQueryAttention_14456859918344
// MI455X (gfx1250) — compile-verified, matrix operand order checked by reading, not run on hardware
//
#include <hip/hip_runtime.h>

#define EMBED 2048
#define TSEQ  2048
#define BATCH 2
#define NQH   32
#define NKVH  8
#define HD    64
#define KVDIM (NKVH * HD)       /* 512  */
#define MROWS (BATCH * TSEQ)    /* 4096 */

// Padded LDS row stride for a 32-element (64B) bf16 row: 64B data + 48B pad.
// 112 = 7*16 keeps b128 alignment; 112/4 = 28 banks => 16 rows hit 16 distinct
// 4-bank groups (minimal 2-pass full-wave ds_load_b128).
#define ROWB 112

typedef __bf16 bf16;
typedef __attribute__((ext_vector_type(16))) __bf16 v16bf;
typedef __attribute__((ext_vector_type(8)))  float  v8f;
typedef __attribute__((ext_vector_type(4)))  unsigned int u32x4;
typedef __attribute__((ext_vector_type(8)))  int i32x8;
typedef __attribute__((ext_vector_type(4)))  int i32x4;

union V8F { v8f v; float f[8]; };

#if defined(__HIP_DEVICE_COMPILE__) && __has_builtin(__builtin_amdgcn_tensor_load_to_lds)
#define USE_TDM 1
#else
#define USE_TDM 0
#endif

// Load one 16x32 bf16 fragment (A layout: lane = row, two 16B chunks per lane).
// Lanes 0-15 (g=0): elems 0..7 -> K=0..7,  elems 8..15 -> K=16..23
// Lanes 16-31 (g=1): elems 0..7 -> K=8..15, elems 8..15 -> K=24..31
__device__ __forceinline__ v16bf load_frag_k32(const bf16* rowptr) {
    const int g = (threadIdx.x & 31) >> 4;
    union { v16bf v; uint4 u[2]; } r;
    r.u[0] = *reinterpret_cast<const uint4*>(rowptr + g * 8);
    r.u[1] = *reinterpret_cast<const uint4*>(rowptr + g * 8 + 16);
    return r.v;
}

__device__ __forceinline__ const bf16* lds_row(const char* base, int r) {
    return (const bf16*)(base + r * ROWB);
}

__device__ __forceinline__ v8f wmma_bf16(v16bf a, v16bf b, v8f c) {
    return __builtin_amdgcn_wmma_f32_16x16x32_bf16(
        false, a, false, b, (short)0, c, false, false);
}

#if USE_TDM
// Issue one TDM 2D tile load: tile_d1 rows x 32 bf16 -> LDS at lds_off,
// padded to ROWB-byte rows via the TDM pad feature.
// 6-arg builtin form: (g0 u32x4, g1 i32x8, g2 i32x4, g3 i32x4, i32x8, cpol)
__device__ __forceinline__ void tdm_load_2d(unsigned lds_off, const bf16* gptr,
                                            int tensor_d0, int tensor_d1,
                                            int tile_d1, int stride0) {
    unsigned long long ga = (unsigned long long)(const void*)gptr;
    u32x4 g0;
    g0[0] = 1u;                                   // count=1, no gather
    g0[1] = lds_off;                              // LDS byte address
    g0[2] = (unsigned)(ga & 0xFFFFFFFFu);         // global addr [31:0]
    g0[3] = (unsigned)((ga >> 32) & 0x1FFFFFFu)   // global addr [56:32]
          | (2u << 30);                           // type = 2 (image)
    i32x8 g1;
    g1[0] = (1 << 16)        // data_size = 2 bytes
          | (1 << 20)        // pad_enable
          | (3 << 22)        // pad_interval code 3 -> every 16 DWORDs (64B row)
          | (11 << 25);      // pad_amount  code 11 -> 12 DWORDs (48B)
    g1[1] = (tensor_d0 & 0xFFFF) << 16;                        // dim0[15:0]
    g1[2] = (int)(((unsigned)tensor_d0 >> 16)                  // dim0[31:16]
          | (((unsigned)tensor_d1 & 0xFFFFu) << 16));          // dim1[15:0]
    g1[3] = (int)(((unsigned)tensor_d1 >> 16)                  // dim1[31:16]
          | (32u << 16));                                      // tile_dim0 = 32
    g1[4] = tile_d1 & 0xFFFF;                                  // tile_dim1
    g1[5] = stride0;                                           // dim0 stride lo
    g1[6] = 0;
    g1[7] = 0;
    i32x4 z4 = {0, 0, 0, 0};
    i32x8 z8 = {0, 0, 0, 0, 0, 0, 0, 0};
    __builtin_amdgcn_tensor_load_to_lds(g0, g1, z4, z4, z8, 0);
}
#endif

// ---------------- conversion kernels ----------------
__global__ void k_convert(const float* __restrict__ src, bf16* __restrict__ dst, int n) {
    int i = blockIdx.x * blockDim.x + threadIdx.x;
    if (i < n) dst[i] = (bf16)src[i];
}

// Wt[n*K + k] = W[k*N + n]
__global__ void k_transpose_convert(const float* __restrict__ W, bf16* __restrict__ Wt,
                                    int K, int N) {
    int i = blockIdx.x * blockDim.x + threadIdx.x;
    if (i >= K * N) return;
    int n = i / K;
    int k = i - n * K;
    Wt[i] = (bf16)W[(long)k * N + n];
}

// ---------------- TDM-staged bf16 GEMM ----------------
// Block tile 128x64, 8 waves of 32x32 (2x2 WMMA accumulators).
// out = A[M,K] * Bt[N,K]^T + bias
// mode 0: bf16 row-major [M,N]
// mode 1: bf16 V-transpose scatter -> Vt[B][NKVH][HD][TSEQ]
// mode 2: f32 row-major [M,N]
__global__ void __launch_bounds__(256)
k_gemm(const bf16* __restrict__ A, int lda,
       const bf16* __restrict__ Bt, int ldb,
       const float* __restrict__ bias,
       void* __restrict__ outp,
       int M, int N, int K, int mode) {
    __shared__ __align__(16) char sA[2][128 * ROWB + 64];
    __shared__ __align__(16) char sB[2][64 * ROWB + 64];

    const int lane = threadIdx.x & 31;
    const int g  = lane >> 4;
    const int ln = lane & 15;
    const int w  = threadIdx.x >> 5;
    const int wm = w & 3;          // 4 wave-rows  -> 128
    const int wn = w >> 2;         // 2 wave-cols  -> 64

    const int nTiles = N >> 6;     // 64-wide N tiles
    const int mt = blockIdx.x / nTiles;
    const int nt = blockIdx.x - mt * nTiles;
    const int m0 = mt * 128, n0 = nt * 64;

    const int S = K >> 5;          // number of 32-wide K slices

#if USE_TDM
    const bool issuer = (w == 0);
    const unsigned offA0 = (unsigned)(unsigned long long)(void*)&sA[0][0];
    const unsigned offA1 = (unsigned)(unsigned long long)(void*)&sA[1][0];
    const unsigned offB0 = (unsigned)(unsigned long long)(void*)&sB[0][0];
    const unsigned offB1 = (unsigned)(unsigned long long)(void*)&sB[1][0];
    if (issuer) {
        tdm_load_2d(offA0, A  + (long)m0 * lda, K, M, 128, lda);
        tdm_load_2d(offB0, Bt + (long)n0 * ldb, K, N, 64,  ldb);
    }
#endif

    v8f c00 = {0.f,0.f,0.f,0.f,0.f,0.f,0.f,0.f};
    v8f c01 = c00, c10 = c00, c11 = c00;

    for (int s = 0; s < S; ++s) {
        const int buf = s & 1;
#if USE_TDM
        if (issuer) {
            if (s + 1 < S) {
                const int k1 = (s + 1) << 5;
                tdm_load_2d(buf ? offA0 : offA1, A  + (long)m0 * lda + k1, K, M, 128, lda);
                tdm_load_2d(buf ? offB0 : offB1, Bt + (long)n0 * ldb + k1, K, N, 64,  ldb);
                __builtin_amdgcn_s_wait_tensorcnt(2);  // slice s pair complete
            } else {
                __builtin_amdgcn_s_wait_tensorcnt(0);
            }
        }
#else
        {   // cooperative staged copy into the same padded layout
            const int k0 = s << 5;
            for (int idx = threadIdx.x; idx < 128 * 4; idx += 256) {
                const int r = idx >> 2, c = idx & 3;
                *(uint4*)(sA[buf] + r * ROWB + c * 16) =
                    *(const uint4*)(A + (long)(m0 + r) * lda + k0 + c * 8);
            }
            for (int idx = threadIdx.x; idx < 64 * 4; idx += 256) {
                const int r = idx >> 2, c = idx & 3;
                *(uint4*)(sB[buf] + r * ROWB + c * 16) =
                    *(const uint4*)(Bt + (long)(n0 + r) * ldb + k0 + c * 8);
            }
        }
#endif
        __syncthreads();

        v16bf a0 = load_frag_k32(lds_row(sA[buf], wm * 32 + ln));
        v16bf a1 = load_frag_k32(lds_row(sA[buf], wm * 32 + 16 + ln));
        v16bf b0 = load_frag_k32(lds_row(sB[buf], wn * 32 + ln));
        v16bf b1 = load_frag_k32(lds_row(sB[buf], wn * 32 + 16 + ln));
        c00 = wmma_bf16(a0, b0, c00);
        c01 = wmma_bf16(a0, b1, c01);
        c10 = wmma_bf16(a1, b0, c10);
        c11 = wmma_bf16(a1, b1, c11);

        __syncthreads();
    }

    V8F cc[2][2];
    cc[0][0].v = c00; cc[0][1].v = c01; cc[1][0].v = c10; cc[1][1].v = c11;

    const int mw = m0 + wm * 32;
    const int nw = n0 + wn * 32;
    for (int si = 0; si < 2; ++si) {
        for (int sj = 0; sj < 2; ++sj) {
            const int ncol = nw + sj * 16 + ln;
            const float bval = bias[ncol];
            for (int i = 0; i < 8; ++i) {
                const int m = mw + si * 16 + i + 8 * g;
                const float val = cc[si][sj].f[i] + bval;
                if (mode == 0) {
                    ((bf16*)outp)[(long)m * N + ncol] = (bf16)val;
                } else if (mode == 1) {
                    const int bb = m / TSEQ, t = m - bb * TSEQ;
                    const int hk = ncol >> 6, d = ncol & 63;
                    ((bf16*)outp)[(((long)(bb * NKVH + hk)) * HD + d) * TSEQ + t] = (bf16)val;
                } else {
                    ((float*)outp)[(long)m * N + ncol] = val;
                }
            }
        }
    }
}

// ---------------- flash attention: 1 wave per (b, h, 16-query tile) ----------------
__global__ void k_attn(const bf16* __restrict__ Q,
                       const bf16* __restrict__ Kb,
                       const bf16* __restrict__ Vt,
                       bf16* __restrict__ O) {
    __shared__ __align__(16) bf16 Plds[16 * 32];

    const int lane = threadIdx.x & 31;
    const int g  = lane >> 4;
    const int ln = lane & 15;

    int id = blockIdx.x;            // B * NQH * (T/16)
    const int qt = id & 127;        // T/16 == 128
    id >>= 7;
    const int h = id & 31;
    const int b = id >> 5;
    const int hk = h >> 2;          // N_REP = 4
    const int q0 = qt * 16;

    const bf16* qrow = Q + (long)(b * TSEQ + q0 + ln) * EMBED + h * HD;
    const v16bf aq0 = load_frag_k32(qrow);
    const v16bf aq1 = load_frag_k32(qrow + 32);

    const bf16* Khead = Kb + (long)b * TSEQ * KVDIM + hk * HD;
    const bf16* Vhead = Vt + (long)(b * NKVH + hk) * HD * TSEQ;

    float mrow[8], lrow[8];
    V8F acc[4];
    for (int i = 0; i < 8; ++i) { mrow[i] = -__builtin_inff(); lrow[i] = 0.f; }
    for (int dt = 0; dt < 4; ++dt)
        for (int i = 0; i < 8; ++i) acc[dt].f[i] = 0.f;

    const int lastkt = qt >> 1;     // 32-key tiles; causal bound
    for (int kt = 0; kt <= lastkt; ++kt) {
        const int kbase = kt * 32;

        V8F s[2];
        for (int st = 0; st < 2; ++st) {
            const bf16* krow = Khead + (long)(kbase + st * 16 + ln) * KVDIM;
            v16bf bk0 = load_frag_k32(krow);
            v16bf bk1 = load_frag_k32(krow + 32);
            v8f c = {0.f,0.f,0.f,0.f,0.f,0.f,0.f,0.f};
            c = wmma_bf16(aq0, bk0, c);
            c = wmma_bf16(aq1, bk1, c);
            s[st].v = c;
        }

        const float scale = 0.125f;   // 1/sqrt(64)
        if (kt == lastkt) {
            for (int st = 0; st < 2; ++st)
                for (int i = 0; i < 8; ++i) {
                    const int qidx = q0 + i + 8 * g;
                    const int kidx = kbase + st * 16 + ln;
                    s[st].f[i] = (kidx > qidx) ? -__builtin_inff() : s[st].f[i] * scale;
                }
        } else {
            for (int st = 0; st < 2; ++st)
                for (int i = 0; i < 8; ++i) s[st].f[i] *= scale;
        }

        for (int i = 0; i < 8; ++i) {
            float t = fmaxf(s[0].f[i], s[1].f[i]);
            t = fmaxf(t, __shfl_xor(t, 1, 32));
            t = fmaxf(t, __shfl_xor(t, 2, 32));
            t = fmaxf(t, __shfl_xor(t, 4, 32));
            t = fmaxf(t, __shfl_xor(t, 8, 32));
            const float mn = fmaxf(mrow[i], t);
            const float alpha = __expf(mrow[i] - mn);
            const float p0 = __expf(s[0].f[i] - mn);
            const float p1 = __expf(s[1].f[i] - mn);
            s[0].f[i] = p0;
            s[1].f[i] = p1;
            float rs = p0 + p1;
            rs += __shfl_xor(rs, 1, 32);
            rs += __shfl_xor(rs, 2, 32);
            rs += __shfl_xor(rs, 4, 32);
            rs += __shfl_xor(rs, 8, 32);
            lrow[i] = lrow[i] * alpha + rs;
            mrow[i] = mn;
            for (int dt = 0; dt < 4; ++dt) acc[dt].f[i] *= alpha;
        }

        for (int st = 0; st < 2; ++st)
            for (int i = 0; i < 8; ++i)
                Plds[(i + 8 * g) * 32 + st * 16 + ln] = (bf16)s[st].f[i];
        __builtin_amdgcn_wave_barrier();
        asm volatile("s_wait_dscnt 0" ::: "memory");
        const v16bf ap = load_frag_k32(Plds + ln * 32);
        __builtin_amdgcn_wave_barrier();

        for (int dt = 0; dt < 4; ++dt) {
            v16bf bv = load_frag_k32(Vhead + (long)(dt * 16 + ln) * TSEQ + kbase);
            acc[dt].v = wmma_bf16(ap, bv, acc[dt].v);
        }
    }

    bf16* obase = O + (long)(b * TSEQ + q0) * EMBED + h * HD;
    for (int i = 0; i < 8; ++i) {
        const int m = i + 8 * g;
        const float inv = 1.f / lrow[i];
        for (int dt = 0; dt < 4; ++dt)
            obase[(long)m * EMBED + dt * 16 + ln] = (bf16)(acc[dt].f[i] * inv);
    }
}

// ---------------- host launch ----------------
extern "C" void kernel_launch(void* const* d_in, const int* in_sizes, int n_in,
                              void* d_out, int out_size, void* d_ws, size_t ws_size,
                              hipStream_t stream) {
    const float* x  = (const float*)d_in[0];
    /* d_in[1] = mask: causal mask applied analytically in k_attn */
    const float* Wq = (const float*)d_in[2];
    const float* bq = (const float*)d_in[3];
    const float* Wk = (const float*)d_in[4];
    const float* bk = (const float*)d_in[5];
    const float* Wv = (const float*)d_in[6];
    const float* bv = (const float*)d_in[7];
    const float* Wo = (const float*)d_in[8];
    const float* bo = (const float*)d_in[9];
    float* out = (float*)d_out;

    char* ws = (char*)d_ws;
    size_t off = 0;
    auto alloc_bf = [&](size_t elems) {
        bf16* p = (bf16*)(ws + off);
        off += ((elems * sizeof(bf16) + 255) & ~(size_t)255);
        return p;
    };
    bf16* xbf = alloc_bf((size_t)MROWS * EMBED);
    bf16* WqT = alloc_bf((size_t)EMBED * EMBED);
    bf16* WkT = alloc_bf((size_t)KVDIM * EMBED);
    bf16* WvT = alloc_bf((size_t)KVDIM * EMBED);
    bf16* WoT = alloc_bf((size_t)EMBED * EMBED);
    bf16* Qb  = alloc_bf((size_t)MROWS * EMBED);
    bf16* Kc  = alloc_bf((size_t)MROWS * KVDIM);
    bf16* Vt  = alloc_bf((size_t)BATCH * KVDIM * TSEQ);
    bf16* Ob  = alloc_bf((size_t)MROWS * EMBED);

    // 1) precision conversion + weight transpose
    {
        int n = MROWS * EMBED;
        k_convert<<<(n + 255) / 256, 256, 0, stream>>>(x, xbf, n);
        k_transpose_convert<<<(EMBED * EMBED + 255) / 256, 256, 0, stream>>>(Wq, WqT, EMBED, EMBED);
        k_transpose_convert<<<(EMBED * KVDIM + 255) / 256, 256, 0, stream>>>(Wk, WkT, EMBED, KVDIM);
        k_transpose_convert<<<(EMBED * KVDIM + 255) / 256, 256, 0, stream>>>(Wv, WvT, EMBED, KVDIM);
        k_transpose_convert<<<(EMBED * EMBED + 255) / 256, 256, 0, stream>>>(Wo, WoT, EMBED, EMBED);
    }

    // 2) QKV projections (TDM-staged bf16 WMMA GEMMs)
    {
        const int blkQ = (MROWS / 128) * (EMBED / 64);   // 1024
        const int blkK = (MROWS / 128) * (KVDIM / 64);   // 256
        k_gemm<<<blkQ, 256, 0, stream>>>(xbf, EMBED, WqT, EMBED, bq, Qb,
                                         MROWS, EMBED, EMBED, 0);
        k_gemm<<<blkK, 256, 0, stream>>>(xbf, EMBED, WkT, EMBED, bk, Kc,
                                         MROWS, KVDIM, EMBED, 0);
        k_gemm<<<blkK, 256, 0, stream>>>(xbf, EMBED, WvT, EMBED, bv, Vt,
                                         MROWS, KVDIM, EMBED, 1);
    }

    // 3) causal GQA flash attention
    k_attn<<<BATCH * NQH * (TSEQ / 16), 32, 0, stream>>>(Qb, Kc, Vt, Ob);

    // 4) output projection (f32 out)
    {
        const int blkO = (MROWS / 128) * (EMBED / 64);
        k_gemm<<<blkO, 256, 0, stream>>>(Ob, EMBED, WoT, EMBED, bo, out,
                                         MROWS, EMBED, EMBED, 2);
    }
}
